// MultiHeadAttention_89223650607839
// MI455X (gfx1250) — matrix operand order repaired and checked by reading, not compiled, not run on hardware
//
#include <hip/hip_runtime.h>
#include <hip/hip_bf16.h>

// ---------------------------------------------------------------------------
// MHA forward for MI455X (gfx1250, wave32, WMMA bf16 16x16x32, f32 accum).
// B=4, T=2048, C=1024, H=16, Dh=64.  softmax scale (and log2(e)) folded into Q.
// GEMMs: LDS-tiled 128x64, double-buffered via global_load_async_to_lds_b128.
// ---------------------------------------------------------------------------

typedef __attribute__((ext_vector_type(16))) __bf16       v16bf;
typedef __attribute__((ext_vector_type(8)))  float        v8f;
typedef __attribute__((ext_vector_type(4)))  unsigned int v4u;

#define B_SZ   4
#define T_SZ   2048
#define C_SZ   1024
#define H_SZ   16
#define DH_SZ  64
#define QSCALE (0.125f * 1.4426950408889634f)   // SCALE * log2(e)

#define ALD 72   // padded LDS row stride (elements) for A tiles
#define BLD 72   // padded LDS row stride (elements) for B tiles

union FragU {
    v16bf v;
    v4u   q[2];
    unsigned int u[8];
};

__device__ __forceinline__ unsigned short f2bf(float f) {
    union { float f; unsigned int u; } x; x.f = f;
    unsigned int r = x.u + 0x7FFFu + ((x.u >> 16) & 1u);   // round-nearest-even
    return (unsigned short)(r >> 16);
}

__device__ __forceinline__ unsigned lds_off(const void* p) {
    // generic pointer to LDS: low 32 bits are the LDS address (aperture rule)
    return (unsigned)(size_t)p;
}

// async DMA: global -> LDS, 16 bytes per lane, tracked by ASYNCcnt
#define ASYNC_B128(ldsdst, gsrc)                                             \
    asm volatile("global_load_async_to_lds_b128 %0, %1, off"                 \
                 :: "v"(ldsdst), "v"((unsigned long long)(size_t)(gsrc))     \
                 : "memory")
#define WAIT_ASYNC_6()  asm volatile("s_wait_asynccnt 0x6" ::: "memory")
#define WAIT_ASYNC_0()  asm volatile("s_wait_asynccnt 0x0" ::: "memory")

// A fragment (16x32, M=lane&15): row-major src, chunks at kb+hi*16 / +8
__device__ __forceinline__ v16bf ldA(const unsigned short* base, int ld,
                                     int row, int kb, int hi) {
    FragU f;
    const unsigned short* p = base + (size_t)row * ld + kb + hi * 16;
    f.q[0] = *(const v4u*)(p);
    f.q[1] = *(const v4u*)(p + 8);
    return f.v;
}

// B fragment (32x16, N=lane&15): N-major src, chunks at kb+hi*16 / +8
__device__ __forceinline__ v16bf ldB(const unsigned short* base, int ld,
                                     int col, int kb, int hi) {
    FragU f;
    const unsigned short* p = base + (size_t)col * ld + kb + hi * 16;
    f.q[0] = *(const v4u*)(p);
    f.q[1] = *(const v4u*)(p + 8);
    return f.v;
}

#define WMMA_BF16(a, b, c) \
    __builtin_amdgcn_wmma_f32_16x16x32_bf16(false, (a), false, (b), (short)0, (c), false, false)

// --------------------------- conversion kernels ----------------------------

__global__ void __launch_bounds__(256) cvt_f32_bf16(const float* __restrict__ src,
                                                    unsigned short* __restrict__ dst, int n) {
    int i = blockIdx.x * blockDim.x + threadIdx.x;
    if (i < n) dst[i] = f2bf(src[i]);
}

// src: K x N row-major f32  ->  dst: N x K row-major bf16 (N-major for B frags)
__global__ void __launch_bounds__(256) cvt_transpose(const float* __restrict__ src,
                                                     unsigned short* __restrict__ dst,
                                                     int N, int K) {
    int i = blockIdx.x * blockDim.x + threadIdx.x;
    if (i < N * K) {
        int nn = i / K, kk = i - nn * K;
        dst[i] = f2bf(src[(size_t)kk * N + nn]);
    }
}

// ---------------------------------------------------------------------------
// LDS-tiled GEMM core (shared by qkv_gemm / out_gemm via macro):
//   block = 256 threads = 8 waves; block tile M=128, N=64, K-step 64;
//   double-buffered async staging; wave w computes rows w*16..w*16+15.
// Produces v8f acc[4] per wave (cols n0 + s*16 + lane15).
// ---------------------------------------------------------------------------
#define GEMM_CORE(Aglob, Bglob)                                                   \
    __shared__ unsigned short shA[2][128][ALD];                                   \
    __shared__ unsigned short shB[2][64][BLD];                                    \
    const int lane = threadIdx.x & 31, l15 = lane & 15, hi = lane >> 4;           \
    const int tid = threadIdx.x;                                                  \
    const int wrow = (threadIdx.x >> 5) * 16;                                     \
    v8f acc[4] = {};                                                              \
    /* stage issue: 4 A-chunks + 2 B-chunks of 16B per thread */                  \
    auto issue = [&](int ks, int buf) {                                           \
        _Pragma("unroll")                                                         \
        for (int i = 0; i < 4; ++i) {                                             \
            int c = tid + 256 * i;                                                \
            int row = c >> 3, col = (c & 7) * 8;                                  \
            ASYNC_B128(lds_off(&shA[buf][row][col]),                              \
                       (Aglob) + (size_t)(m0 + row) * C_SZ + ks * 64 + col);      \
        }                                                                         \
        _Pragma("unroll")                                                         \
        for (int i = 0; i < 2; ++i) {                                             \
            int c = tid + 256 * i;                                                \
            int cn = c >> 3, kk = (c & 7) * 8;                                    \
            ASYNC_B128(lds_off(&shB[buf][cn][kk]),                                \
                       (Bglob) + (size_t)(n0 + cn) * C_SZ + ks * 64 + kk);        \
        }                                                                         \
    };                                                                            \
    issue(0, 0);                                                                  \
    for (int ks = 0; ks < C_SZ / 64; ++ks) {                                      \
        int buf = ks & 1;                                                         \
        if (ks + 1 < C_SZ / 64) { issue(ks + 1, buf ^ 1); WAIT_ASYNC_6(); }       \
        else                    { WAIT_ASYNC_0(); }                               \
        __syncthreads();                                                          \
        const unsigned short* Ab = &shA[buf][wrow][0];                            \
        const unsigned short* Bb = &shB[buf][0][0];                               \
        v16bf a0 = ldA(Ab, ALD, l15, 0,  hi);                                     \
        v16bf a1 = ldA(Ab, ALD, l15, 32, hi);                                     \
        v16bf b0[4], b1[4];                                                       \
        _Pragma("unroll")                                                         \
        for (int s = 0; s < 4; ++s) {                                             \
            b0[s] = ldB(Bb, BLD, s * 16 + l15, 0,  hi);                           \
            b1[s] = ldB(Bb, BLD, s * 16 + l15, 32, hi);                           \
        }                                                                         \
        _Pragma("unroll")                                                         \
        for (int s = 0; s < 4; ++s) {                                             \
            acc[s] = WMMA_BF16(a0, b0[s], acc[s]);                                \
            acc[s] = WMMA_BF16(a1, b1[s], acc[s]);                                \
        }                                                                         \
        __syncthreads();                                                          \
    }

// ------------------------------ QKV GEMM -----------------------------------
// X_bf (8192x1024) @ Wt (3072x1024, N-major) + b -> scatter to Q*QSCALE / K / V^T
__global__ void __launch_bounds__(256) qkv_gemm(const unsigned short* __restrict__ xb,
                                                const unsigned short* __restrict__ wt,
                                                const float* __restrict__ bias,
                                                unsigned short* __restrict__ qo,
                                                unsigned short* __restrict__ ko,
                                                unsigned short* __restrict__ vto) {
    const int mt = blockIdx.x / 48, nt = blockIdx.x - mt * 48;
    const int m0 = mt * 128, n0 = nt * 64;
    GEMM_CORE(xb, wt)

    const int which = n0 >> 10;            // 0=Q 1=K 2=V (uniform per wave)
    const int cb = n0 & 1023;
    #pragma unroll
    for (int s = 0; s < 4; ++s) {
        int n = n0 + s * 16 + l15;
        float bi = bias[n];
        int c = cb + s * 16 + l15, hh = c >> 6, dd = c & 63;
        #pragma unroll
        for (int r = 0; r < 8; ++r) {
            int tok = m0 + wrow + r + hi * 8;
            int bb = tok >> 11, tt = tok & 2047;
            float v = acc[s][r] + bi;
            if (which == 0)
                qo[(((size_t)(bb * H_SZ + hh)) * T_SZ + tt) * DH_SZ + dd] = f2bf(v * QSCALE);
            else if (which == 1)
                ko[(((size_t)(bb * H_SZ + hh)) * T_SZ + tt) * DH_SZ + dd] = f2bf(v);
            else
                vto[(((size_t)(bb * H_SZ + hh)) * DH_SZ + dd) * T_SZ + tt] = f2bf(v);
        }
    }
}

// ---------------------------- output projection ----------------------------
__global__ void __launch_bounds__(256) out_gemm(const unsigned short* __restrict__ ab,
                                                const unsigned short* __restrict__ wt,
                                                const float* __restrict__ bias,
                                                float* __restrict__ out) {
    const int mt = blockIdx.x >> 4, nt = blockIdx.x & 15;
    const int m0 = mt * 128, n0 = nt * 64;
    GEMM_CORE(ab, wt)

    #pragma unroll
    for (int s = 0; s < 4; ++s) {
        int n = n0 + s * 16 + l15;
        float bi = bias[n];
        #pragma unroll
        for (int r = 0; r < 8; ++r)
            out[(size_t)(m0 + wrow + r + hi * 8) * C_SZ + n] = acc[s][r] + bi;
    }
}

// --------------------------- flash attention -------------------------------
// Per wave: 16 queries. Transposed form: S^T = K_blk @ Q^T ; O^T = V^T @ P^T.
// Lane owns one query column -> softmax state is lane-local (+1 shfl_xor(16)).
// All K/V fragments for an iteration are issued before the WMMAs; V latency
// is hidden under the softmax / P^T-pack VALU work.
__global__ void __launch_bounds__(128) flash_attn(const unsigned short* __restrict__ qb,
                                                  const unsigned short* __restrict__ kb_,
                                                  const unsigned short* __restrict__ vt,
                                                  unsigned short* __restrict__ attn) {
    const int lane = threadIdx.x & 31, l15 = lane & 15, hi = lane >> 4;
    const int w = threadIdx.x >> 5;
    const int bh = blockIdx.x >> 5;          // 0..63
    const int qblk = blockIdx.x & 31;        // 0..31
    const int t0 = qblk * 64 + w * 16;

    const unsigned short* qbase = qb  + ((size_t)bh * T_SZ + t0) * DH_SZ;
    const unsigned short* kbase = kb_ + (size_t)bh * T_SZ * DH_SZ;
    const unsigned short* vbase = vt  + (size_t)bh * DH_SZ * T_SZ;

    // Q^T held as B fragments for the whole kernel (score scale pre-folded)
    v16bf qf0 = ldB(qbase, DH_SZ, l15, 0,  hi);
    v16bf qf1 = ldB(qbase, DH_SZ, l15, 32, hi);

    v8f accO[4] = {};                        // O^T: 4 tiles of 16d x 16q
    float m_run = -1e30f, l_run = 0.0f;

    for (int j = 0; j < T_SZ / 64; ++j) {
        // ---- issue ALL loads for this iteration up front ----
        v16bf ka[8];
        #pragma unroll
        for (int t = 0; t < 4; ++t) {
            const unsigned short* kp = kbase + (size_t)(j * 64 + t * 16) * DH_SZ;
            ka[2 * t]     = ldA(kp, DH_SZ, l15, 0,  hi);
            ka[2 * t + 1] = ldA(kp, DH_SZ, l15, 32, hi);
        }
        v16bf va[8];
        #pragma unroll
        for (int m = 0; m < 4; ++m) {
            const unsigned short* vp = vbase + (size_t)(m * 16) * T_SZ;
            #pragma unroll
            for (int g = 0; g < 2; ++g)
                va[2 * m + g] = ldA(vp, T_SZ, l15, j * 64 + g * 32, hi);
        }
        if (j + 1 < T_SZ / 64) {
            __builtin_prefetch(kbase + (size_t)(j + 1) * 64 * DH_SZ, 0, 0);
            __builtin_prefetch(vbase + (size_t)(j + 1) * 64, 0, 0);
        }

        // ---- S^T (64 keys x 16 queries) = K_blk @ Q^T ----
        v8f S[4] = {};
        #pragma unroll
        for (int t = 0; t < 4; ++t) {
            S[t] = WMMA_BF16(ka[2 * t],     qf0, S[t]);
            S[t] = WMMA_BF16(ka[2 * t + 1], qf1, S[t]);
        }

        // ---- online softmax (log2 domain, scale already in Q) ----
        float mloc = -1e30f;
        #pragma unroll
        for (int t = 0; t < 4; ++t)
            #pragma unroll
            for (int r = 0; r < 8; ++r) mloc = fmaxf(mloc, S[t][r]);
        mloc = fmaxf(mloc, __shfl_xor(mloc, 16, 32));
        float m_new = fmaxf(m_run, mloc);

        float sloc = 0.0f;
        #pragma unroll
        for (int t = 0; t < 4; ++t)
            #pragma unroll
            for (int r = 0; r < 8; ++r) {
                float p = exp2f(S[t][r] - m_new);
                S[t][r] = p;
                sloc += p;
            }
        sloc += __shfl_xor(sloc, 16, 32);
        float alpha = exp2f(m_run - m_new);
        l_run = l_run * alpha + sloc;
        m_run = m_new;
        #pragma unroll
        for (int m = 0; m < 4; ++m)
            #pragma unroll
            for (int r = 0; r < 8; ++r) accO[m][r] *= alpha;

        // ---- P^T -> B fragments, in-register (one cross-half shfl per row) ----
        FragU pf[2];
        #pragma unroll
        for (int g = 0; g < 2; ++g) {
            float lo[8], hv[8];
            #pragma unroll
            for (int r = 0; r < 8; ++r) {
                float own = hi ? S[2 * g + 1][r] : S[2 * g][r];     // tile 2g+hi, rows 8hi+r
                float snd = hi ? S[2 * g][r]     : S[2 * g + 1][r]; // partner needs this
                float oth = __shfl_xor(snd, 16, 32);                // tile 2g+hi, other rows
                lo[r] = hi ? oth : own;
                hv[r] = hi ? own : oth;
            }
            #pragma unroll
            for (int jj = 0; jj < 4; ++jj)
                pf[g].u[jj] = (unsigned int)f2bf(lo[2 * jj]) |
                              ((unsigned int)f2bf(lo[2 * jj + 1]) << 16);
            #pragma unroll
            for (int jj = 0; jj < 4; ++jj)
                pf[g].u[4 + jj] = (unsigned int)f2bf(hv[2 * jj]) |
                                  ((unsigned int)f2bf(hv[2 * jj + 1]) << 16);
        }

        // ---- O^T += V^T @ P^T  (fragments already resident) ----
        #pragma unroll
        for (int m = 0; m < 4; ++m)
            #pragma unroll
            for (int g = 0; g < 2; ++g)
                accO[m] = WMMA_BF16(va[2 * m + g], pf[g].v, accO[m]);
    }

    // ---- normalize and scatter bf16 attention output, token-major ----
    float inv = 1.0f / l_run;
    int bb = bh >> 4, hh = bh & 15;
    unsigned short* obase = attn + ((size_t)(bb * T_SZ + t0 + l15)) * C_SZ + hh * DH_SZ;
    #pragma unroll
    for (int m = 0; m < 4; ++m)
        #pragma unroll
        for (int r = 0; r < 8; ++r)
            obase[m * 16 + r + hi * 8] = f2bf(accO[m][r] * inv);
}

// ------------------------------- launcher ----------------------------------
extern "C" void kernel_launch(void* const* d_in, const int* in_sizes, int n_in,
                              void* d_out, int out_size, void* d_ws, size_t ws_size,
                              hipStream_t stream) {
    const float* x     = (const float*)d_in[0];
    const float* w_qkv = (const float*)d_in[1];
    const float* b_qkv = (const float*)d_in[2];
    const float* w_out = (const float*)d_in[3];
    const float* b_out = (const float*)d_in[4];
    float* out = (float*)d_out;

    char* ws = (char*)d_ws;
    unsigned short* xb    = (unsigned short*)(ws);                       // 16 MB
    unsigned short* wqkvt = (unsigned short*)(ws + 16777216);            //  6 MB
    unsigned short* woutt = (unsigned short*)(ws + 23068672);            //  2 MB
    unsigned short* qo    = (unsigned short*)(ws + 25165824);            // 16 MB
    unsigned short* ko    = (unsigned short*)(ws + 41943040);            // 16 MB
    unsigned short* vto   = (unsigned short*)(ws + 58720256);            // 16 MB (V^T)
    unsigned short* attnb = (unsigned short*)(ws + 75497472);            // 16 MB

    int nx = (int)((size_t)B_SZ * T_SZ * C_SZ);   // 8388608
    cvt_f32_bf16<<<(nx + 255) / 256, 256, 0, stream>>>(x, xb, nx);
    cvt_transpose<<<(3 * C_SZ * C_SZ + 255) / 256, 256, 0, stream>>>(w_qkv, wqkvt, 3 * C_SZ, C_SZ);
    cvt_transpose<<<(C_SZ * C_SZ + 255) / 256, 256, 0, stream>>>(w_out, woutt, C_SZ, C_SZ);

    // QKV: 64 M-blocks (128 rows) x 48 N-blocks (64 cols) = 3072 blocks
    qkv_gemm<<<3072, 256, 0, stream>>>(xb, wqkvt, b_qkv, qo, ko, vto);

    // 64 (b,h) x 32 query-blocks of 64 = 2048 blocks, 4 waves each
    flash_attn<<<2048, 128, 0, stream>>>(qo, ko, vto, attnb);

    // out-proj: 64 M-blocks x 16 N-blocks = 1024 blocks
    out_gemm<<<1024, 256, 0, stream>>>(attnb, woutt, b_out, out);
}
